// SDPA_22024592294625
// MI455X (gfx1250) — matrix operand order repaired and checked by reading, not compiled, not run on hardware
//
#include <hip/hip_runtime.h>

// ---------------------------------------------------------------------------
// Head-axis-softmax attention for MI455X (gfx1250, wave32, WMMA + TDM).
//
// scores[b,n,q,k] = (Q[b,q,n,:] . K[b,k,n,:]) / 8
// w = softmax over n (16 heads)           <- separable over k!
// out[b,n,q,d] = sum_k w[b,n,q,k] * V[b,k,n,d]
//
// bf16 WMMA 16x16x32 for both GEMMs, f32 accumulate. One workgroup =
// 16 queries x 16 heads (one wave per head), streaming 64 blocks of 32 keys.
// K blocks are staged by the Tensor Data Mover into a pad-strided LDS tile;
// V is VALU-transposed into [h][d][k] bf16 for contiguous PV fragments.
// ---------------------------------------------------------------------------

#define BB 2
#define SS 2048
#define NH 16
#define DD 64
#define QT 16
#define KB 32
#define NKB (SS / KB)

typedef __attribute__((ext_vector_type(16))) __bf16 v16bf;
typedef __attribute__((ext_vector_type(8)))  __bf16 v8bf;
typedef __attribute__((ext_vector_type(8)))  float  v8f;
typedef __attribute__((ext_vector_type(4)))  float  v4f;
typedef __attribute__((ext_vector_type(4)))  unsigned int u32x4;
typedef __attribute__((ext_vector_type(4)))  int i32x4;
typedef __attribute__((ext_vector_type(8)))  int i32x8;

// LDS map (bytes):
//  [0)       qlds  bf16 [16h][16m][64d]          32768  (pre-loop only; aliased)
//  [0)       slds  f32  [16q][32k][20 pad]       40960
//  [40960)   kf32  f32  TDM tile [16h][32k][68dw]139264 (64 d + 4 pad dwords/row)
//  [180224)  vlds  bf16 [16h][64d x 40][32k]      82176
//  [262400)  wf    bf16 [16h][32lane][16elem]     16384
//  total 278784
#define S_OFF  0
#define KF_OFF 40960
#define V_OFF  180224
#define W_OFF  262400
#define SMEM_BYTES 278784
#define KROW 68        // dwords per (h,k) row of K in LDS (64 + 4 pad)
#define VHSTRIDE 2568  // elements per head in vlds (64*40 + 8 pad)
#define VDSTRIDE 40    // elements per d-row in vlds (32 + 8 pad)

__device__ __forceinline__ v16bf load16(const __bf16* p0, const __bf16* p1) {
  v8bf a = *(const v8bf*)p0;
  v8bf b = *(const v8bf*)p1;
  v16bf r;
#pragma unroll
  for (int i = 0; i < 8; ++i) { r[i] = a[i]; r[i + 8] = b[i]; }
  return r;
}

// 16 contiguous f32 from LDS -> bf16 fragment
__device__ __forceinline__ v16bf cvt16(const float* p) {
  v4f a = *(const v4f*)(p + 0);
  v4f b = *(const v4f*)(p + 4);
  v4f c = *(const v4f*)(p + 8);
  v4f d = *(const v4f*)(p + 12);
  v16bf r;
  r[0]  = (__bf16)a.x; r[1]  = (__bf16)a.y; r[2]  = (__bf16)a.z; r[3]  = (__bf16)a.w;
  r[4]  = (__bf16)b.x; r[5]  = (__bf16)b.y; r[6]  = (__bf16)b.z; r[7]  = (__bf16)b.w;
  r[8]  = (__bf16)c.x; r[9]  = (__bf16)c.y; r[10] = (__bf16)c.z; r[11] = (__bf16)c.w;
  r[12] = (__bf16)d.x; r[13] = (__bf16)d.y; r[14] = (__bf16)d.z; r[15] = (__bf16)d.w;
  return r;
}

// TDM: copy one K block [32k][16h][64d] f32 (contiguous 128KB) into LDS as a
// 3D tile X=d(64, contig), Y=k(32, stride 1024), Z=h(16, stride 64) -> LDS
// order [h][k][d], with 4 pad dwords after every 64-dword d-row (row = 68 dw).
__device__ __forceinline__ void tdm_load_k(const float* gsrc, unsigned lds_addr) {
  unsigned long long ga = (unsigned long long)(const void*)gsrc;
  u32x4 g0;
  g0[0] = 1u;                                   // count=1, user descriptor
  g0[1] = lds_addr;                             // lds_addr [63:32]
  g0[2] = (unsigned)(ga & 0xffffffffu);         // global_addr [95:64]
  g0[3] = (unsigned)((ga >> 32) & 0x1ffffffu)   // global_addr [120:96]
          | (2u << 30);                         // type=2 ("image")
  i32x8 g1;
  g1[0] = (int)((2u << 16)       // data_size = 4B
              | (1u << 20)       // pad_enable
              | (5u << 22)       // pad_interval: every 64 dwords
              | (3u << 25));     // pad_amount: 4 dwords
  g1[1] = (int)(64u << 16);      // tensor_dim0 = 64   (X = d)
  g1[2] = (int)(32u << 16);      // tensor_dim1 = 32   (Y = k)
  g1[3] = (int)(64u << 16);      // tile_dim0   = 64
  g1[4] = (int)(32u | (16u << 16)); // tile_dim1 = 32, tile_dim2 = 16
  g1[5] = 1024;                  // tensor_dim0_stride = 1024 (k step, elems)
  g1[6] = (int)(64u << 16);      // tensor_dim1_stride = 64   (h step, elems)
  g1[7] = 0;
  i32x4 g2;
  g2[0] = 16;                    // tensor_dim2 = 16 (Z = h)
  g2[1] = 0; g2[2] = 0; g2[3] = 0;
  i32x4 g3 = {0, 0, 0, 0};
  i32x8 g4 = {0, 0, 0, 0, 0, 0, 0, 0};  // unused trailing group (6-arg form)
  __builtin_amdgcn_tensor_load_to_lds(g0, g1, g2, g3, g4, 0);
}

__global__ __launch_bounds__(512, 1)
void sdpa_headsoftmax(const float* __restrict__ Qg, const float* __restrict__ Kg,
                      const float* __restrict__ Vg, const int* __restrict__ Mg,
                      float* __restrict__ Out) {
  extern __shared__ char smem[];
  __bf16* qlds = (__bf16*)(smem);          // pre-loop alias of slds region
  float*  slds = (float*)(smem + S_OFF);
  float*  kf32 = (float*)(smem + KF_OFF);
  __bf16* vlds = (__bf16*)(smem + V_OFF);
  __bf16* wf   = (__bf16*)(smem + W_OFF);
  const unsigned kf_lds = (unsigned)(unsigned long long)(const void*)(smem + KF_OFF);

  const int tid  = threadIdx.x;       // 0..511
  const int lane = tid & 31;
  const int wave = tid >> 5;          // == head index (0..15)
  const int lg   = lane >> 4;         // lane group (0/1)
  const int ln   = lane & 15;
  const int b    = blockIdx.y;
  const int q0   = blockIdx.x * QT;

  // ---------------- stage Q tile: f32 -> bf16 (fold 1/sqrt(64)) -----------
  {
    const float* src = Qg + (size_t)(b * SS + q0) * NH * DD + (size_t)tid * 32;
    const int m  = tid >> 5;
    const int h  = (tid >> 1) & 15;
    const int d0 = (tid & 1) * 32;
    alignas(16) __bf16 tmp[32];
#pragma unroll
    for (int i = 0; i < 32; i += 4) {
      v4f v = *(const v4f*)(src + i);
      tmp[i + 0] = (__bf16)(v.x * 0.125f);
      tmp[i + 1] = (__bf16)(v.y * 0.125f);
      tmp[i + 2] = (__bf16)(v.z * 0.125f);
      tmp[i + 3] = (__bf16)(v.w * 0.125f);
    }
    __bf16* dst = qlds + (h * 16 + m) * 64 + d0;
#pragma unroll
    for (int i = 0; i < 32; i += 8) *(v8bf*)(dst + i) = *(const v8bf*)(tmp + i);
  }
  __syncthreads();

  // Persistent Q A-fragments (16x32 bf16 per d-chunk).
  // A-layout: lane ln holds row m=ln; element e -> d = chunk*32+lg*16+e
  v16bf qa0, qa1;
  {
    const __bf16* base = qlds + (wave * 16 + ln) * 64 + lg * 16;
    qa0 = load16(base + 0,  base + 8);
    qa1 = load16(base + 32, base + 40);
  }

  v8f acc[4] = {{}, {}, {}, {}};   // out[16q x 64d] f32, 4 chunks of 16 d

  for (int kb = 0; kb < NKB; ++kb) {
    const int k0 = kb * KB;
    const float* kgp = Kg + (size_t)(b * SS + k0) * NH * DD;
    const float* vgp = Vg + (size_t)(b * SS + k0) * NH * DD;

    // ---- TDM: async-stage this K block into LDS (wave 0 issues) ----------
    if (wave == 0) tdm_load_k(kgp, kf_lds);

    if (kb + 1 < NKB) {  // pull next block toward L2 (global_prefetch_b8)
      __builtin_prefetch(kgp + NH * DD * KB + tid * 64, 0, 1);
      __builtin_prefetch(vgp + NH * DD * KB + tid * 64, 0, 1);
    }

    // ---- stage V transposed: [h][d][k] bf16 (overlaps TDM in flight) -----
    {
      const int k = tid >> 4;      // 0..31
      const int h = tid & 15;
      const float* src = vgp + (size_t)(k * NH + h) * DD;
      __bf16* dst = vlds + h * VHSTRIDE + k;
#pragma unroll
      for (int d = 0; d < 64; d += 4) {
        v4f a = *(const v4f*)(src + d);
        dst[(d + 0) * VDSTRIDE] = (__bf16)a.x;
        dst[(d + 1) * VDSTRIDE] = (__bf16)a.y;
        dst[(d + 2) * VDSTRIDE] = (__bf16)a.z;
        dst[(d + 3) * VDSTRIDE] = (__bf16)a.w;
      }
    }
    if (wave == 0) __builtin_amdgcn_s_wait_tensorcnt(0);
    __syncthreads();

    // ---------------- scores: S_h = Q_h (16x64) x K_h^T (64 x 32) --------
#pragma unroll
    for (int ks = 0; ks < 2; ++ks) {
      // K lds row (h*32 + k) * 68 dwords; B-layout: col n = key = ln,
      // element e -> d = chunk*32 + lg*16 + e (contiguous f32, cvt to bf16)
      const float* kbase = kf32 + (wave * 32 + ks * 16 + ln) * KROW + lg * 16;
      v16bf b0 = cvt16(kbase + 0);    // d-chunk 0
      v16bf b1 = cvt16(kbase + 32);   // d-chunk 1
      v8f s = {};
      s = __builtin_amdgcn_wmma_f32_16x16x32_bf16(false, qa0, false, b0,
                                                  (short)0, s, false, false);
      s = __builtin_amdgcn_wmma_f32_16x16x32_bf16(false, qa1, false, b1,
                                                  (short)0, s, false, false);
      // D-layout: VGPR j -> row m = j + 8*lg ; col = ln
#pragma unroll
      for (int j = 0; j < 8; ++j) {
        const int m = j + 8 * lg;
        slds[(m * KB + ks * 16 + ln) * 20 + wave] = s[j];
      }
    }
    __syncthreads();

    // ---------------- softmax over the 16 heads, one (q,k) per thread ----
    {
      const int q  = tid >> 5;
      const int kk = tid & 31;
      const float* sp = slds + (q * KB + kk) * 20;
      v4f s0 = *(const v4f*)(sp + 0);
      v4f s1 = *(const v4f*)(sp + 4);
      v4f s2 = *(const v4f*)(sp + 8);
      v4f s3 = *(const v4f*)(sp + 12);
      float sv[16] = {s0.x, s0.y, s0.z, s0.w, s1.x, s1.y, s1.z, s1.w,
                      s2.x, s2.y, s2.z, s2.w, s3.x, s3.y, s3.z, s3.w};
      const int mv = Mg[(size_t)(b * SS + (q0 + q)) * SS + (k0 + kk)];
      if (mv == 0) {
#pragma unroll
        for (int h = 0; h < 16; ++h) sv[h] = -__builtin_inff();
      }
      float mx = sv[0];
#pragma unroll
      for (int h = 1; h < 16; ++h) mx = __builtin_fmaxf(mx, sv[h]);
      float sum = 0.f;
#pragma unroll
      for (int h = 0; h < 16; ++h) { sv[h] = __expf(sv[h] - mx); sum += sv[h]; }
      const float inv = __builtin_amdgcn_rcpf(sum);
      // scatter into per-head WMMA A-fragment layout:
      // (m=q, k=kk) -> lane q + 16*bit3(kk), element (kk/16)*8 + kk%8
      const int lt = q + 16 * ((kk >> 3) & 1);
      const int et = (kk >> 4) * 8 + (kk & 7);
#pragma unroll
      for (int h = 0; h < 16; ++h)
        wf[(h * 32 + lt) * 16 + et] = (__bf16)(sv[h] * inv);
    }
    __syncthreads();

    // ---------------- PV: out_h += W_h (16x32) x V_h (32x64) -------------
    {
      const __bf16* wbase = wf + (wave * 32 + lane) * 16;
      v16bf wa = load16(wbase, wbase + 8);
#pragma unroll
      for (int c = 0; c < 4; ++c) {
        // B-layout: col n = d = c*16 + ln ; element e -> key = (e/8)*16 + lg*8 + e%8
        const __bf16* vp = vlds + wave * VHSTRIDE + (c * 16 + ln) * VDSTRIDE + lg * 8;
        v16bf vb = load16(vp, vp + 16);
        acc[c] = __builtin_amdgcn_wmma_f32_16x16x32_bf16(false, wa, false, vb,
                                                         (short)0, acc[c], false, false);
      }
    }
    __syncthreads();   // protect kf32/vlds before next iteration's staging
  }

  // ---------------- epilogue: out[b][h][q][d] f32 -------------------------
  {
    float* ob = Out + ((size_t)(b * NH + wave) * SS + q0) * DD;
#pragma unroll
    for (int c = 0; c < 4; ++c)
#pragma unroll
      for (int j = 0; j < 8; ++j) {
        const int m = j + 8 * lg;
        ob[(size_t)m * DD + c * 16 + ln] = acc[c][j];
      }
  }
}

extern "C" void kernel_launch(void* const* d_in, const int* in_sizes, int n_in,
                              void* d_out, int out_size, void* d_ws, size_t ws_size,
                              hipStream_t stream) {
  (void)in_sizes; (void)n_in; (void)out_size; (void)d_ws; (void)ws_size;
  const float* Q = (const float*)d_in[0];
  const float* K = (const float*)d_in[1];
  const float* V = (const float*)d_in[2];
  const int*   M = (const int*)d_in[3];
  float* O = (float*)d_out;
  dim3 grid(SS / QT, BB);
  sdpa_headsoftmax<<<grid, dim3(512), SMEM_BYTES, stream>>>(Q, K, V, M, O);
}
